// PGCN_16415365005353
// MI455X (gfx1250) — matrix operand order repaired and checked by reading, not compiled, not run on hardware
//
#include <hip/hip_runtime.h>
#include <hip/hip_bf16.h>
#include <math.h>

#define N_NODES 100000
#define N_EDGES 1600000
#define FEAT    512
#define HID     128
#define CLS     40
#define CLSP    48      // CLS padded to 3 WMMA tiles
#define LN_EPS  1e-5f

typedef __attribute__((ext_vector_type(16))) __bf16 v16bf;
typedef __attribute__((ext_vector_type(8)))  float  v8f;

union AFrag { v16bf v; unsigned int u[8]; };
union BFrag { v16bf v; uint4 q[2]; };

__device__ __forceinline__ float wave_sum(float v) {
#pragma unroll
  for (int o = 16; o > 0; o >>= 1) v += __shfl_xor(v, o, 32);
  return v;
}
__device__ __forceinline__ float wave_max(float v) {
#pragma unroll
  for (int o = 16; o > 0; o >>= 1) v = fmaxf(v, __shfl_xor(v, o, 32));
  return v;
}

// ---------- weight packing: transposed bf16 so B fragments are contiguous ----------
__global__ void k_pack_w1(const float* __restrict__ W1, __bf16* __restrict__ w1t) {
  int i = blockIdx.x * blockDim.x + threadIdx.x;       // over HID*FEAT
  if (i >= HID * FEAT) return;
  int n = i / FEAT, k = i % FEAT;                      // w1t[n][k] = W1[k][n]
  w1t[i] = (__bf16)W1[k * HID + n];
}

__global__ void k_pack_w2(const float* __restrict__ W2, __bf16* __restrict__ w2t) {
  int i = blockIdx.x * blockDim.x + threadIdx.x;       // over CLSP*HID
  if (i >= CLSP * HID) return;
  int c = i / HID, k = i % HID;                        // w2t[c][k] = W2[k][c], zero-pad
  w2t[i] = (c < CLS) ? (__bf16)W2[k * CLS + c] : (__bf16)0.0f;
}

// ---------- degree / normalization ----------
__global__ void k_degree(const int* __restrict__ dst, const float* __restrict__ ew,
                         float* __restrict__ deg) {
  int e = blockIdx.x * blockDim.x + threadIdx.x;
  if (e < N_EDGES) atomicAdd(&deg[dst[e]], ew[e]);
}

__global__ void k_dinv(const float* __restrict__ deg, float* __restrict__ dinv,
                       float* __restrict__ dinv2) {
  int n = blockIdx.x * blockDim.x + threadIdx.x;
  if (n >= N_NODES) return;
  float r = rsqrtf(deg[n] + 1.0f);                     // self-loop adds 1
  dinv[n] = r;
  dinv2[n] = r * r;
}

__global__ void k_norm(const int* __restrict__ src, const int* __restrict__ dst,
                       const float* __restrict__ ew, const float* __restrict__ dinv,
                       float* __restrict__ norm) {
  int e = blockIdx.x * blockDim.x + threadIdx.x;
  if (e < N_EDGES) norm[e] = dinv[src[e]] * ew[e] * dinv[dst[e]];
}

// ---------- fused LayerNorm + GEMM1 (bf16 WMMA), 16 rows x 128 cols per block ----------
#define ASTR 520   // 512 + 8 bf16 pad -> row stride 1040B, 16-way bank-conflict free
__global__ __launch_bounds__(256) void k_ln_gemm1(
    const float* __restrict__ x, const float* __restrict__ gamma,
    const float* __restrict__ beta, const __bf16* __restrict__ w1t,
    float* __restrict__ h1) {
  __shared__ __bf16 lds_a[16 * ASTR];
  const int lane = threadIdx.x & 31;
  const int wave = threadIdx.x >> 5;                   // 0..7
  const int rowbase = blockIdx.x * 16;

  // LayerNorm: each wave normalizes 2 rows, writes bf16 A-tile into LDS
#pragma unroll
  for (int rr = 0; rr < 2; ++rr) {
    const int r = wave * 2 + rr;
    const float* xr = x + (size_t)(rowbase + r) * FEAT;
    float v[16];
    float s = 0.f;
#pragma unroll
    for (int j = 0; j < 16; ++j) { v[j] = xr[lane + 32 * j]; s += v[j]; }
    const float mu = wave_sum(s) * (1.0f / FEAT);
    float q = 0.f;
#pragma unroll
    for (int j = 0; j < 16; ++j) { float d = v[j] - mu; q += d * d; }
    const float rs = rsqrtf(wave_sum(q) * (1.0f / FEAT) + LN_EPS);
#pragma unroll
    for (int j = 0; j < 16; ++j) {
      const int c = lane + 32 * j;
      lds_a[r * ASTR + c] = (__bf16)((v[j] - mu) * rs * gamma[c] + beta[c]);
    }
  }
  __syncthreads();

  // GEMM: each wave owns a 16x16 tile at column wave*16
  const int m  = lane & 15;
  const int kh = lane >> 4;
  const int ncol = wave * 16 + (lane & 15);
  const __bf16* bp = w1t + (size_t)ncol * FEAT;
  v8f acc = {};
#pragma unroll
  for (int kk = 0; kk < FEAT / 32; ++kk) {
    AFrag a; BFrag b;
#pragma unroll
    for (int j = 0; j < 8; ++j) {   // ISA 16-bit A layout: K pairs per VGPR
      const int Kj = kk * 32 + ((j < 4) ? 0 : 16) + kh * 8 + (j & 3) * 2;
      a.u[j] = *(const unsigned int*)&lds_a[m * ASTR + Kj];
    }
    const int kb = kk * 32 + kh * 8; // B layout matches A: K 8h..8h+7, then 16+8h..16+8h+7
    b.q[0] = *(const uint4*)(bp + kb);
    b.q[1] = *(const uint4*)(bp + kb + 16);
    acc = __builtin_amdgcn_wmma_f32_16x16x32_bf16(false, a.v, false, b.v,
                                                  (short)0, acc, false, false);
  }
#pragma unroll
  for (int r = 0; r < 8; ++r) {     // C/D layout: VGPR r -> M=r (lanes 0-15), r+8 (16-31)
    const int row = rowbase + r + kh * 8;
    h1[(size_t)row * HID + ncol] = acc[r];
  }
}

// ---------- edge scatter, 128 features: one wave per edge, float4 gather ----------
__global__ __launch_bounds__(256) void k_agg128(
    const int* __restrict__ src, const int* __restrict__ dst,
    const float* __restrict__ norm, const float* __restrict__ h1,
    float* __restrict__ agg1) {
  const int e = blockIdx.x * 8 + (threadIdx.x >> 5);
  if (e >= N_EDGES) return;
  const int lane = threadIdx.x & 31;
  const int s = src[e], d = dst[e];
  const float w = norm[e];
  const float4 hv = *(const float4*)(h1 + (size_t)s * HID + lane * 4);
  float* ap = agg1 + (size_t)d * HID + lane * 4;
  atomicAdd(ap + 0, hv.x * w);
  atomicAdd(ap + 1, hv.y * w);
  atomicAdd(ap + 2, hv.z * w);
  atomicAdd(ap + 3, hv.w * w);
}

// ---------- self-loop + bias for layer 1 ----------
__global__ void k_selfloop1(const float* __restrict__ agg1, const float* __restrict__ h1,
                            const float* __restrict__ dinv2, const float* __restrict__ b1,
                            float* __restrict__ x1) {
  const int i = blockIdx.x * blockDim.x + threadIdx.x;
  if (i >= N_NODES * HID) return;
  x1[i] = agg1[i] + h1[i] * dinv2[i >> 7] + b1[i & (HID - 1)];
}

// ---------- GEMM2: 16 rows x 48 cols per block (3 compute waves) ----------
#define ASTR2 136  // 128 + 8 pad
__global__ __launch_bounds__(128) void k_gemm2(
    const float* __restrict__ x1, const __bf16* __restrict__ w2t,
    float* __restrict__ h2) {
  __shared__ __bf16 lds_a[16 * ASTR2];
  const int tid = threadIdx.x;
  const int rowbase = blockIdx.x * 16;
#pragma unroll
  for (int j = 0; j < 16; ++j)
    lds_a[j * ASTR2 + tid] = (__bf16)x1[(size_t)(rowbase + j) * HID + tid];
  __syncthreads();
  const int wave = tid >> 5;
  if (wave >= 3) return;
  const int lane = tid & 31;
  const int m = lane & 15, kh = lane >> 4;
  const int ncol = wave * 16 + (lane & 15);
  const __bf16* bp = w2t + (size_t)ncol * HID;
  v8f acc = {};
#pragma unroll
  for (int kk = 0; kk < HID / 32; ++kk) {
    AFrag a; BFrag b;
#pragma unroll
    for (int j = 0; j < 8; ++j) {
      const int Kj = kk * 32 + ((j < 4) ? 0 : 16) + kh * 8 + (j & 3) * 2;
      a.u[j] = *(const unsigned int*)&lds_a[m * ASTR2 + Kj];
    }
    const int kb = kk * 32 + kh * 8;
    b.q[0] = *(const uint4*)(bp + kb);
    b.q[1] = *(const uint4*)(bp + kb + 16);
    acc = __builtin_amdgcn_wmma_f32_16x16x32_bf16(false, a.v, false, b.v,
                                                  (short)0, acc, false, false);
  }
#pragma unroll
  for (int r = 0; r < 8; ++r)
    h2[(size_t)(rowbase + r + kh * 8) * CLSP + ncol] = acc[r];
}

// ---------- edge scatter, 40 features ----------
__global__ __launch_bounds__(256) void k_agg40(
    const int* __restrict__ src, const int* __restrict__ dst,
    const float* __restrict__ norm, const float* __restrict__ h2,
    float* __restrict__ agg2) {
  const int e = blockIdx.x * 8 + (threadIdx.x >> 5);
  if (e >= N_EDGES) return;
  const int lane = threadIdx.x & 31;
  const int s = src[e], d = dst[e];
  const float w = norm[e];
  atomicAdd(agg2 + (size_t)d * CLS + lane, h2[(size_t)s * CLSP + lane] * w);
  if (lane < CLS - 32)
    atomicAdd(agg2 + (size_t)d * CLS + 32 + lane,
              h2[(size_t)s * CLSP + 32 + lane] * w);
}

// ---------- self-loop + bias + log_softmax: one wave per row ----------
__global__ __launch_bounds__(256) void k_final(
    const float* __restrict__ agg2, const float* __restrict__ h2,
    const float* __restrict__ dinv2, const float* __restrict__ b2,
    float* __restrict__ out) {
  const int row = blockIdx.x * 8 + (threadIdx.x >> 5);
  if (row >= N_NODES) return;
  const int lane = threadIdx.x & 31;
  const float di = dinv2[row];
  float va = agg2[(size_t)row * CLS + lane] + h2[(size_t)row * CLSP + lane] * di + b2[lane];
  float vb = -3.402823466e38f;
  if (lane < 8)
    vb = agg2[(size_t)row * CLS + 32 + lane] +
         h2[(size_t)row * CLSP + 32 + lane] * di + b2[32 + lane];
  const float mx = wave_max(fmaxf(va, vb));
  float s = __expf(va - mx) + ((lane < 8) ? __expf(vb - mx) : 0.f);
  s = wave_sum(s);
  const float lse = __logf(s) + mx;
  out[(size_t)row * CLS + lane] = va - lse;
  if (lane < 8) out[(size_t)row * CLS + 32 + lane] = vb - lse;
}

extern "C" void kernel_launch(void* const* d_in, const int* in_sizes, int n_in,
                              void* d_out, int out_size, void* d_ws, size_t ws_size,
                              hipStream_t stream) {
  (void)in_sizes; (void)n_in; (void)out_size; (void)ws_size;
  const float* x     = (const float*)d_in[0];
  const int*   eidx  = (const int*)d_in[1];   // int32 (JAX default x64-disabled)
  const float* ew    = (const float*)d_in[2];
  const float* gamma = (const float*)d_in[3];
  const float* beta  = (const float*)d_in[4];
  const float* W1    = (const float*)d_in[5];
  const float* b1    = (const float*)d_in[6];
  const float* W2    = (const float*)d_in[7];
  const float* b2    = (const float*)d_in[8];
  const int* src = eidx;
  const int* dst = eidx + N_EDGES;
  float* out = (float*)d_out;

  char* ws = (char*)d_ws;
  size_t off = 0;
  auto alloc = [&](size_t bytes) -> void* {
    off = (off + 255) & ~(size_t)255;
    void* p = ws + off;
    off += bytes;
    return p;
  };
  __bf16* w1t  = (__bf16*)alloc((size_t)HID * FEAT * 2);
  __bf16* w2t  = (__bf16*)alloc((size_t)CLSP * HID * 2);
  float*  deg   = (float*)alloc((size_t)N_NODES * 4);
  float*  dinv  = (float*)alloc((size_t)N_NODES * 4);
  float*  dinv2 = (float*)alloc((size_t)N_NODES * 4);
  float*  norm  = (float*)alloc((size_t)N_EDGES * 4);
  float*  h1    = (float*)alloc((size_t)N_NODES * HID * 4);
  float*  agg1  = (float*)alloc((size_t)N_NODES * HID * 4);
  float*  x1    = (float*)alloc((size_t)N_NODES * HID * 4);
  float*  h2    = (float*)alloc((size_t)N_NODES * CLSP * 4);
  float*  agg2  = (float*)alloc((size_t)N_NODES * CLS * 4);

  k_pack_w1<<<(HID * FEAT + 255) / 256, 256, 0, stream>>>(W1, w1t);
  k_pack_w2<<<(CLSP * HID + 255) / 256, 256, 0, stream>>>(W2, w2t);

  hipMemsetAsync(deg, 0, (size_t)N_NODES * 4, stream);
  k_degree<<<N_EDGES / 256, 256, 0, stream>>>(dst, ew, deg);
  k_dinv<<<(N_NODES + 255) / 256, 256, 0, stream>>>(deg, dinv, dinv2);
  k_norm<<<N_EDGES / 256, 256, 0, stream>>>(src, dst, ew, dinv, norm);

  // layer 1
  k_ln_gemm1<<<N_NODES / 16, 256, 0, stream>>>(x, gamma, beta, w1t, h1);
  hipMemsetAsync(agg1, 0, (size_t)N_NODES * HID * 4, stream);
  k_agg128<<<N_EDGES / 8, 256, 0, stream>>>(src, dst, norm, h1, agg1);
  k_selfloop1<<<(N_NODES * HID + 255) / 256, 256, 0, stream>>>(agg1, h1, dinv2, b1, x1);

  // layer 2
  k_gemm2<<<N_NODES / 16, 128, 0, stream>>>(x1, w2t, h2);
  hipMemsetAsync(agg2, 0, (size_t)N_NODES * CLS * 4, stream);
  k_agg40<<<N_EDGES / 8, 256, 0, stream>>>(src, dst, norm, h2, agg2);
  k_final<<<N_NODES / 8, 256, 0, stream>>>(agg2, h2, dinv2, b2, out);
}
